// CausalTransformer_35510789603978
// MI455X (gfx1250) — matrix operand order repaired and checked by reading, not compiled, not run on hardware
//
#include <hip/hip_runtime.h>
#include <hip/hip_bf16.h>

// ---------------------------------------------------------------------------
// Types
// ---------------------------------------------------------------------------
typedef __bf16 bf16;
typedef __attribute__((ext_vector_type(16))) __bf16 v16bf;
typedef __attribute__((ext_vector_type(8)))  __bf16 v8bf;
typedef __attribute__((ext_vector_type(8)))  float  v8f;

#define NEG_BIG (-1.0e9f)

__device__ __forceinline__ unsigned short f2bfbits(float x) {
  union { float f; unsigned u; } c; c.f = x;
  unsigned r = c.u + 0x7FFFu + ((c.u >> 16) & 1u);
  return (unsigned short)(r >> 16);
}
__device__ __forceinline__ bf16 f2bf(float x) {
  union { unsigned short s; bf16 b; } u; u.s = f2bfbits(x); return u.b;
}
__device__ __forceinline__ v8f zero8() {
  v8f z = {0.f, 0.f, 0.f, 0.f, 0.f, 0.f, 0.f, 0.f};
  return z;
}

// ---------------------------------------------------------------------------
// WMMA tile loaders (wave32, 16x16x32 bf16, contiguous k order shared by A and B)
// A (16x32, MxK): lanes 0-15 row M=lane hold K 0..15; lanes 16-31
// hold K 16..31 (2 bf16 per VGPR).
// ---------------------------------------------------------------------------
__device__ __forceinline__ v16bf load_a(const bf16* __restrict__ A, long lda,
                                        int m0, int k0, int lane) {
  const int half = lane >> 4, r = lane & 15;
  const bf16* p = A + (long)(m0 + r) * lda + k0 + (half << 4);
  v8bf lo = *(const v8bf*)(p);
  v8bf hi = *(const v8bf*)(p + 8);
  return __builtin_shufflevector(lo, hi, 0, 1, 2, 3, 4, 5, 6, 7,
                                 8, 9, 10, 11, 12, 13, 14, 15);
}

// B (32x16, KxN) built from W stored row-major NxK (B[k][n] = W[n][k]):
// lane n (lane&15) holds column n; lanes 0-15 K=0..15, lanes 16-31 K=16..31.
__device__ __forceinline__ v16bf load_b_nk(const bf16* __restrict__ W, long ldw,
                                           int n0, int k0, int lane, int Nmax) {
  const int kh = lane >> 4, r = lane & 15;
  int n = n0 + r; if (n >= Nmax) n = Nmax - 1;   // clamp: padded cols unused
  return *(const v16bf*)(W + (long)n * ldw + k0 + (kh << 4));
}

// B (32x16, KxN) from a row-major KxN matrix (used for V in attention).
__device__ __forceinline__ v16bf load_b_kn(const bf16* __restrict__ Bm, long ldb,
                                           int k0, int n0, int lane) {
  const int kh = lane >> 4, r = lane & 15;
  const bf16* p = Bm + (long)(k0 + (kh << 4)) * ldb + n0 + r;
  union { v16bf v; bf16 e[16]; } u;
#pragma unroll
  for (int i = 0; i < 16; ++i) u.e[i] = p[(long)i * ldb];
  return u.v;
}

// A tile from LDS where bf16 P values live in the low 16 bits of f32 slots
// (element stride 2 ushorts), row stride 1024 elements.
__device__ __forceinline__ v16bf load_a_p(const unsigned short* __restrict__ pp,
                                          int k0, int lane) {
  const int half = lane >> 4, r = lane & 15;
  const unsigned short* base = pp + ((long)(r << 10) + k0 + (half << 4)) * 2;
  union { v16bf v; unsigned short s[16]; } u;
#pragma unroll
  for (int i = 0; i < 8; ++i) u.s[i] = base[2 * i];
#pragma unroll
  for (int i = 0; i < 8; ++i) u.s[8 + i] = base[2 * (8 + i)];
  return u.v;
}

__device__ __forceinline__ v8f wmma_bf16(v16bf a, v16bf b, v8f c) {
  return __builtin_amdgcn_wmma_f32_16x16x32_bf16(
      /*neg_a=*/false, a, /*neg_b=*/false, b,
      /*c_mod=*/(short)0, c, /*reuse_a=*/false, /*reuse_b=*/false);
}

// ---------------------------------------------------------------------------
// f32 -> bf16 conversion (weights)
// ---------------------------------------------------------------------------
__global__ void cvt_bf16_kernel(const float* __restrict__ in,
                                bf16* __restrict__ out, long n) {
  long i = (long)blockIdx.x * blockDim.x + threadIdx.x;
  long stride = (long)gridDim.x * blockDim.x;
  for (; i < n; i += stride) out[i] = f2bf(in[i]);
}

// ---------------------------------------------------------------------------
// LayerNorm: one block per row; optional f32 and bf16 outputs
// ---------------------------------------------------------------------------
__global__ __launch_bounds__(256) void ln_kernel(
    const float* __restrict__ in, const float* __restrict__ g,
    const float* __restrict__ bta, float* __restrict__ out_f,
    bf16* __restrict__ out_b, int D, float eps) {
  __shared__ float red_s[8], red_q[8];
  const int tid = threadIdx.x;
  const float* x = in + (long)blockIdx.x * D;
  float s = 0.f, q = 0.f;
  for (int i = tid; i < D; i += 256) { float v = x[i]; s += v; q += v * v; }
#pragma unroll
  for (int off = 16; off; off >>= 1) {
    s += __shfl_xor(s, off, 32);
    q += __shfl_xor(q, off, 32);
  }
  if ((tid & 31) == 0) { red_s[tid >> 5] = s; red_q[tid >> 5] = q; }
  __syncthreads();
  float ts = 0.f, tq = 0.f;
#pragma unroll
  for (int i = 0; i < 8; ++i) { ts += red_s[i]; tq += red_q[i]; }
  const float mean = ts / (float)D;
  const float var = tq / (float)D - mean * mean;
  const float rstd = rsqrtf(var + eps);
  for (int i = tid; i < D; i += 256) {
    float y = (x[i] - mean) * rstd * g[i] + bta[i];
    long idx = (long)blockIdx.x * D + i;
    if (out_f) out_f[idx] = y;
    if (out_b) out_b[idx] = f2bf(y);
  }
}

// ---------------------------------------------------------------------------
// WMMA GEMM: out[M,N] = A[M,K](bf16) @ W[N,K]^T(bf16) + bias, with epilogues.
// mode 0: f32 out;  1: f32 out += residual;  2: exact-GELU -> bf16;  3: bf16
// One wave computes a 32x64 tile (2 A tiles x 4 B tiles, 8 accumulators) so
// every B tile is reused across two A tiles: ~21 FLOP/byte from L2.
// 8 waves / block. Requires M % 32 == 0 (true for all GEMMs here: M=2048).
// ---------------------------------------------------------------------------
__global__ __launch_bounds__(256) void gemm_wmma_kernel(
    const bf16* __restrict__ A, const bf16* __restrict__ W,
    const float* __restrict__ bias, const float* __restrict__ res,
    float* __restrict__ out_f, bf16* __restrict__ out_b,
    int M, int N, int K, int lda, int ldw, int ldo, int mode) {
  const int lane = threadIdx.x & 31;
  const int tiles_n = (N + 63) >> 6;
  const int tiles_m = M >> 5;
  const int wid = blockIdx.x * 8 + (threadIdx.x >> 5);
  if (wid >= tiles_m * tiles_n) return;          // wave-uniform guard
  const int tm = wid / tiles_n, tn = wid % tiles_n;
  const int m0 = tm << 5, n0 = tn << 6;

  v8f acc[2][4];
#pragma unroll
  for (int i = 0; i < 2; ++i)
#pragma unroll
    for (int t = 0; t < 4; ++t) acc[i][t] = zero8();

  for (int k0 = 0; k0 < K; k0 += 32) {
    v16bf a0 = load_a(A, lda, m0, k0, lane);
    v16bf a1 = load_a(A, lda, m0 + 16, k0, lane);
    v16bf b0 = load_b_nk(W, ldw, n0, k0, lane, N);
    v16bf b1 = load_b_nk(W, ldw, n0 + 16, k0, lane, N);
    v16bf b2 = load_b_nk(W, ldw, n0 + 32, k0, lane, N);
    v16bf b3 = load_b_nk(W, ldw, n0 + 48, k0, lane, N);
    acc[0][0] = wmma_bf16(a0, b0, acc[0][0]);
    acc[1][0] = wmma_bf16(a1, b0, acc[1][0]);
    acc[0][1] = wmma_bf16(a0, b1, acc[0][1]);
    acc[1][1] = wmma_bf16(a1, b1, acc[1][1]);
    acc[0][2] = wmma_bf16(a0, b2, acc[0][2]);
    acc[1][2] = wmma_bf16(a1, b2, acc[1][2]);
    acc[0][3] = wmma_bf16(a0, b3, acc[0][3]);
    acc[1][3] = wmma_bf16(a1, b3, acc[1][3]);
  }

  const int half = lane >> 4, r = lane & 15;
#pragma unroll
  for (int i = 0; i < 2; ++i) {
#pragma unroll
    for (int t = 0; t < 4; ++t) {
      const int n = n0 + (t << 4) + r;
      if (n >= N) continue;
      const float bv = bias ? bias[n] : 0.f;
#pragma unroll
      for (int v = 0; v < 8; ++v) {
        const int m = m0 + (i << 4) + (half << 3) + v;
        const long idx = (long)m * ldo + n;
        float val = acc[i][t][v] + bv;
        if (mode == 1) val += res[idx];
        if (mode == 2) val = 0.5f * val * (1.f + erff(val * 0.70710678118654752f));
        if (mode >= 2) out_b[idx] = f2bf(val);
        else           out_f[idx] = val;
      }
    }
  }
}

// ---------------------------------------------------------------------------
// Attention: one block per (batch, head, 16-query tile).
//   Phase 1: scores 16x1024 via WMMA -> 64 KB LDS (f32); Q tile hoisted
//   Phase 2: scale + rel-pos + causal bias, row softmax (shfl width-16),
//            P stored as bf16 in-place in the low half of each f32 slot
//   Phase 3: O(16x64) = P(16x1024) @ V(1024x64) via WMMA, 4 waves
// qkv is bf16 [2048 x 3072] (q|k|v); o_out bf16 [2048 x 1024].
// ---------------------------------------------------------------------------
__global__ __launch_bounds__(256) void attn_kernel(
    const bf16* __restrict__ qkv, const float* __restrict__ rel_table,
    bf16* __restrict__ o_out) {
  __shared__ float sc[16 * 1024];                 // exactly 64 KB
  const int tid = threadIdx.x;
  const int lane = tid & 31, w = tid >> 5;
  const int mt = blockIdx.x & 63;
  const int h  = (blockIdx.x >> 6) & 15;
  const int b  = blockIdx.x >> 10;
  const long base = (long)b * 1024 * 3072;
  const bf16* Q  = qkv + base + (long)h * 64;
  const bf16* Kp = qkv + base + 1024 + (long)h * 64;
  const bf16* Vp = qkv + base + 2048 + (long)h * 64;
  const int m0 = mt << 4;
  const int half = lane >> 4, r = lane & 15;

  // Phase 1: scores (Q tile is loop-invariant; load once)
  {
    const v16bf qa0 = load_a(Q + (long)m0 * 3072, 3072, 0, 0, lane);
    const v16bf qa1 = load_a(Q + (long)m0 * 3072, 3072, 0, 32, lane);
    for (int nt = w; nt < 64; nt += 8) {
      v8f acc = zero8();
      v16bf b0 = load_b_nk(Kp, 3072, nt << 4, 0, lane, 1024);
      acc = wmma_bf16(qa0, b0, acc);
      v16bf b1 = load_b_nk(Kp, 3072, nt << 4, 32, lane, 1024);
      acc = wmma_bf16(qa1, b1, acc);
#pragma unroll
      for (int v = 0; v < 8; ++v)
        sc[((half << 3) + v) * 1024 + (nt << 4) + r] = acc[v];
    }
  }
  __syncthreads();

  // Phase 2: bias + softmax (row = tid>>4, 16 lanes per row, 64 cols each)
  {
    const int row = tid >> 4, j = tid & 15;
    const int qg = m0 + row;
    const float* rt = rel_table + h * 257;
    float mx = -3.4e38f;
    for (int c = 0; c < 64; ++c) {
      const int kcol = j + (c << 4);
      float s = sc[row * 1024 + kcol] * 0.125f;
      int d = kcol - qg;
      d = d < -128 ? -128 : (d > 128 ? 128 : d);
      s += rt[d + 128];
      if (kcol > qg) s += NEG_BIG;
      sc[row * 1024 + kcol] = s;
      mx = fmaxf(mx, s);
    }
#pragma unroll
    for (int off = 1; off < 16; off <<= 1) mx = fmaxf(mx, __shfl_xor(mx, off, 16));
    float sum = 0.f;
    for (int c = 0; c < 64; ++c) {
      const int kcol = j + (c << 4);
      float e = __expf(sc[row * 1024 + kcol] - mx);
      sc[row * 1024 + kcol] = e;
      sum += e;
    }
#pragma unroll
    for (int off = 1; off < 16; off <<= 1) sum += __shfl_xor(sum, off, 16);
    const float inv = 1.f / sum;
    unsigned short* ps = (unsigned short*)sc;
    for (int c = 0; c < 64; ++c) {
      const int kcol = j + (c << 4);
      ps[(row * 1024 + kcol) * 2] = f2bfbits(sc[row * 1024 + kcol] * inv);
    }
  }
  __syncthreads();

  // Phase 3: O = P @ V (waves 0..3, one 16-wide slice of head dim each)
  if (w < 4) {
    const int nt = w;
    const unsigned short* pp = (const unsigned short*)sc;
    v8f acc = zero8();
    for (int k0 = 0; k0 < 1024; k0 += 32) {
      v16bf a = load_a_p(pp, k0, lane);
      v16bf bb = load_b_kn(Vp, 3072, k0, nt << 4, lane);
      acc = wmma_bf16(a, bb, acc);
    }
#pragma unroll
    for (int v = 0; v < 8; ++v) {
      const int m = m0 + (half << 3) + v;
      o_out[(long)(b * 1024 + m) * 1024 + h * 64 + (nt << 4) + r] = f2bf(acc[v]);
    }
  }
}

// ---------------------------------------------------------------------------
// Host orchestration
// ---------------------------------------------------------------------------
static inline void launch_cvt(hipStream_t s, const float* in, bf16* out, long n) {
  int blocks = (int)((n + 255) / 256); if (blocks > 4096) blocks = 4096;
  cvt_bf16_kernel<<<blocks, 256, 0, s>>>(in, out, n);
}
static inline void launch_gemm(hipStream_t s, const bf16* A, const bf16* W,
                               const float* bias, const float* res,
                               float* out_f, bf16* out_b,
                               int M, int N, int K, int lda, int ldw, int ldo,
                               int mode) {
  const int tiles = (M >> 5) * ((N + 63) >> 6);
  gemm_wmma_kernel<<<(tiles + 7) / 8, 256, 0, s>>>(A, W, bias, res, out_f,
                                                   out_b, M, N, K, lda, ldw,
                                                   ldo, mode);
}

extern "C" void kernel_launch(void* const* d_in, const int* in_sizes, int n_in,
                              void* d_out, int out_size, void* d_ws,
                              size_t ws_size, hipStream_t stream) {
  (void)in_sizes; (void)n_in; (void)out_size; (void)ws_size;
  const float* x        = (const float*)d_in[0];
  const float* pe_ln1_g = (const float*)d_in[1];
  const float* pe_ln1_b = (const float*)d_in[2];
  const float* pe_w     = (const float*)d_in[3];
  const float* pe_b     = (const float*)d_in[4];
  const float* pe_ln2_g = (const float*)d_in[5];
  const float* pe_ln2_b = (const float*)d_in[6];
  const float* lnA_g    = (const float*)d_in[7];
  const float* lnA_b    = (const float*)d_in[8];
  const float* qkv_w    = (const float*)d_in[9];
  const float* qkv_b    = (const float*)d_in[10];
  const float* out_w    = (const float*)d_in[11];
  const float* out_b    = (const float*)d_in[12];
  const float* lnF_g    = (const float*)d_in[13];
  const float* lnF_b    = (const float*)d_in[14];
  const float* fc1_w    = (const float*)d_in[15];
  const float* fc1_b    = (const float*)d_in[16];
  const float* fc2_w    = (const float*)d_in[17];
  const float* fc2_b    = (const float*)d_in[18];
  const float* rel_tab  = (const float*)d_in[19];
  const float* final_g  = (const float*)d_in[20];
  const float* final_b  = (const float*)d_in[21];
  const float* head_w   = (const float*)d_in[22];
  const float* head_b   = (const float*)d_in[23];
  float* logits = (float*)d_out;

  // Workspace carve-up (all offsets 256-B aligned), ~55.6 MB total
  char* ws = (char*)d_ws;
  bf16*  hbf   = (bf16*)(ws);                    //  5,242,880  (2048x1280 bf16)
  bf16*  wbuf  = (bf16*)(ws + 5242880);          //  8,388,608  (<=4096x1024 bf16)
  bf16*  qkvbf = (bf16*)(ws + 13631488);         // 12,582,912  (2048x3072 bf16)
  bf16*  obf   = (bf16*)(ws + 26214400);         //  4,194,304  (2048x1024 bf16)
  bf16*  fbf   = (bf16*)(ws + 30408704);         // 16,777,216  (2048x4096 bf16)
  float* z     = (float*)(ws + 47185920);        //  8,388,608  (2048x1024 f32)
  float* ztmp  = (float*)fbf;                    // alias: used only pre-layers

  const int TOK = 2048;   // B*S

  // Patch embed: x viewed [2048 x 1280] -> LN1 -> GEMM -> LN2 -> z
  ln_kernel<<<TOK, 256, 0, stream>>>(x, pe_ln1_g, pe_ln1_b, nullptr, hbf,
                                     1280, 1e-6f);
  launch_cvt(stream, pe_w, wbuf, (long)1024 * 1280);
  launch_gemm(stream, hbf, wbuf, pe_b, nullptr, ztmp, nullptr,
              TOK, 1024, 1280, 1280, 1280, 1024, /*mode=*/0);
  ln_kernel<<<TOK, 256, 0, stream>>>(ztmp, pe_ln2_g, pe_ln2_b, z, nullptr,
                                     1024, 1e-6f);

  for (int l = 0; l < 8; ++l) {
    // Attention block
    ln_kernel<<<TOK, 256, 0, stream>>>(z, lnA_g + l * 1024, lnA_b + l * 1024,
                                       nullptr, hbf, 1024, 1e-5f);
    launch_cvt(stream, qkv_w + (long)l * 3072 * 1024, wbuf, (long)3072 * 1024);
    launch_gemm(stream, hbf, wbuf, qkv_b + l * 3072, nullptr, nullptr, qkvbf,
                TOK, 3072, 1024, 1024, 1024, 3072, /*mode=*/3);
    attn_kernel<<<2048, 256, 0, stream>>>(qkvbf, rel_tab, obf);
    launch_cvt(stream, out_w + (long)l * 1024 * 1024, wbuf, (long)1024 * 1024);
    launch_gemm(stream, obf, wbuf, out_b + l * 1024, z, z, nullptr,
                TOK, 1024, 1024, 1024, 1024, 1024, /*mode=*/1);

    // MLP block
    ln_kernel<<<TOK, 256, 0, stream>>>(z, lnF_g + l * 1024, lnF_b + l * 1024,
                                       nullptr, hbf, 1024, 1e-5f);
    launch_cvt(stream, fc1_w + (long)l * 4096 * 1024, wbuf, (long)4096 * 1024);
    launch_gemm(stream, hbf, wbuf, fc1_b + l * 4096, nullptr, nullptr, fbf,
                TOK, 4096, 1024, 1024, 1024, 4096, /*mode=*/2);
    launch_cvt(stream, fc2_w + (long)l * 1024 * 4096, wbuf, (long)1024 * 4096);
    launch_gemm(stream, fbf, wbuf, fc2_b + l * 1024, z, z, nullptr,
                TOK, 1024, 4096, 4096, 4096, 1024, /*mode=*/1);
  }

  // Final LN + head
  ln_kernel<<<TOK, 256, 0, stream>>>(z, final_g, final_b, nullptr, hbf,
                                     1024, 1e-5f);
  launch_cvt(stream, head_w, wbuf, (long)100 * 1024);
  launch_gemm(stream, hbf, wbuf, head_b, nullptr, logits, nullptr,
              TOK, 100, 1024, 1024, 1024, 100, /*mode=*/0);
}
